// VectorQuantize_2808908612134
// MI455X (gfx1250) — matrix operand order repaired and checked by reading, not compiled, not run on hardware
//
#include <hip/hip_runtime.h>

#define DECAY 0.8f
#define OMD   0.2f
#define EPS_  1e-5f

static constexpr int N_ROWS = 16384;   // 8*2048
static constexpr int DIMS   = 256;
static constexpr int NCODE  = 8192;

typedef __attribute__((ext_vector_type(16))) __bf16 v16bf;
typedef __attribute__((ext_vector_type(8)))  float  v8f;

struct alignas(16) U4 { unsigned x, y, z, w; };
struct U4x2 { U4 a, b; };

__device__ __forceinline__ v16bf frag16(U4 a, U4 b) {
  U4x2 t; t.a = a; t.b = b;
  return __builtin_bit_cast(v16bf, t);
}

__device__ __forceinline__ unsigned f2ord(float f) {
  unsigned u = __float_as_uint(f);
  return (u & 0x80000000u) ? ~u : (u | 0x80000000u);
}

__device__ __forceinline__ unsigned long long shfl_xor_u64(unsigned long long v, int m) {
  unsigned lo = (unsigned)v, hi = (unsigned)(v >> 32);
  lo = __shfl_xor(lo, m, 32);
  hi = __shfl_xor(hi, m, 32);
  return ((unsigned long long)hi << 32) | lo;
}

// CDNA5 async global->LDS copy (ASYNCcnt-tracked), GVS addressing:
//   LDS address from the vdst VGPR, global address = SGPR base + VGPR byte offset
__device__ __forceinline__ void async_b128(unsigned lds_byte_off, const void* sbase,
                                           unsigned goff_bytes) {
  asm volatile("global_load_async_to_lds_b128 %0, %1, %2"
               :
               : "v"(lds_byte_off), "v"(goff_bytes), "s"(sbase)
               : "memory");
}

template <int N>
__device__ __forceinline__ void wait_async() {
  asm volatile("s_wait_asynccnt %0" : : "i"(N) : "memory");
}

// ---------------- init workspace ----------------
__global__ void vq_init(float* __restrict__ embed_sum,
                        unsigned long long* __restrict__ minpack,
                        float* __restrict__ counts,
                        float* __restrict__ loss, float* __restrict__ cs_sum) {
  size_t i = (size_t)blockIdx.x * 256 + threadIdx.x;
  if (i < (size_t)DIMS * NCODE) embed_sum[i] = 0.0f;
  if (i < N_ROWS) minpack[i] = 0xFFFFFFFFFFFFFFFFull;
  if (i < NCODE) counts[i] = 0.0f;
  if (i == 0) { *loss = 0.0f; *cs_sum = 0.0f; }
}

// ---------------- f32 -> bf16 hi/lo split (row-major passthrough) ----------------
__global__ void vq_split(const float* __restrict__ src,
                         unsigned short* __restrict__ hi,
                         unsigned short* __restrict__ lo, size_t n) {
  size_t i = (size_t)blockIdx.x * 256 + threadIdx.x;
  if (i >= n) return;
  float x = src[i];
  unsigned u = __float_as_uint(x);
  hi[i] = (unsigned short)(u >> 16);
  float l = x - __uint_as_float(u & 0xFFFF0000u);
  lo[i] = (unsigned short)(__float_as_uint(l) >> 16);
}

// ---------------- embed split + transpose: [d][c] -> [c][d] ----------------
__global__ void vq_split_t(const float* __restrict__ embed,
                           unsigned short* __restrict__ hi,
                           unsigned short* __restrict__ lo) {
  size_t o = (size_t)blockIdx.x * 256 + threadIdx.x;  // o = c*DIMS + d
  int d = (int)(o & (DIMS - 1));
  int c = (int)(o >> 8);
  float x = embed[(size_t)d * NCODE + c];
  unsigned u = __float_as_uint(x);
  hi[o] = (unsigned short)(u >> 16);
  float l = x - __uint_as_float(u & 0xFFFF0000u);
  lo[o] = (unsigned short)(__float_as_uint(l) >> 16);
}

// ---------------- ||e_c||^2 (coalesced: thread = column) ----------------
__global__ void vq_enorm(const float* __restrict__ embed, float* __restrict__ e_norm) {
  int c = blockIdx.x * 256 + threadIdx.x;
  float s = 0.f;
  for (int d = 0; d < DIMS; ++d) { float v = embed[(size_t)d * NCODE + c]; s += v * v; }
  e_norm[c] = s;
}

// ---------------- fused bf16x3 WMMA GEMM + argmin ----------------
// block tile: 128 rows x 256 cols, K-step 32, 8 waves (2x4), each wave 4x4 16x16 tiles.
// Double-buffered LDS filled via global_load_async_to_lds_b128.
//
// LDS layout per buffer (ushort units, pad stride 40 per 32-k row/col):
//   Ah: [0, 5120)  Al: [5120, 10240)  Bh: [10240, 20480)  Bl: [20480, 30720)
static constexpr int LDS_BUF_US = 30720;           // ushorts per buffer
static constexpr int A_LO_US = 5120;
static constexpr int B_HI_US = 10240;
static constexpr int B_LO_US = 20480;

__global__ void __launch_bounds__(256) vq_argmin_gemm(
    const unsigned short* __restrict__ x_hi, const unsigned short* __restrict__ x_lo,
    const unsigned short* __restrict__ e_hi_t, const unsigned short* __restrict__ e_lo_t,
    const float* __restrict__ e_norm, unsigned long long* __restrict__ minpack) {
  __shared__ unsigned short lds[2 * LDS_BUF_US];

  const int row0 = blockIdx.y * 128;
  const int col0 = blockIdx.x * 256;
  const int t = threadIdx.x;
  const int lane = t & 31;
  const int wave = t >> 5;
  const int wm = wave >> 2;   // 0..1  (64 rows each)
  const int wn = wave & 3;    // 0..3  (64 cols each)
  const int ml = lane & 15;
  const int hf = lane >> 4;

  const unsigned lds_base = (unsigned)(size_t)(void*)&lds[0];  // LDS byte offset of array

  // Issue 12 async b128 copies for one k-tile into buffer `buf`.
  auto stage = [&](int buf, int kk) {
    const unsigned bb = lds_base + (unsigned)buf * (LDS_BUF_US * 2);  // byte base
    // A: 128 rows x 32 k, row-major (pad stride 40 ushorts = 80B). 512 chunks of 16B.
#pragma unroll
    for (int rep = 0; rep < 2; ++rep) {
      int id = t + rep * 256;         // 0..511
      int i = id >> 2, j = id & 3;
      unsigned g = (unsigned)((row0 + i) * (DIMS * 2) + kk * 2 + j * 16);
      unsigned l = bb + (unsigned)(i * 80 + j * 16);
      async_b128(l, x_hi, g);
      async_b128(l + A_LO_US * 2, x_lo, g);
    }
    // B: 256 cols x 32 k from code-major e_t (col slice is 64B contiguous).
    {
      int c = t;                       // one column per thread, 4 chunks
#pragma unroll
      for (int j = 0; j < 4; ++j) {
        unsigned g = (unsigned)((col0 + c) * (DIMS * 2) + kk * 2 + j * 16);
        unsigned l = bb + (unsigned)(B_HI_US * 2 + c * 80 + j * 16);
        async_b128(l, e_hi_t, g);
        async_b128(l + (B_LO_US - B_HI_US) * 2, e_lo_t, g);
      }
    }
  };

  v8f zero = {};
  v8f acc[4][4];
#pragma unroll
  for (int a = 0; a < 4; ++a)
#pragma unroll
    for (int b = 0; b < 4; ++b) acc[a][b] = zero;

  stage(0, 0);
  int cur = 0;
  for (int ki = 0; ki < 8; ++ki) {
    if (ki < 7) stage(cur ^ 1, (ki + 1) * 32);
    if (ki < 7) wait_async<12>(); else wait_async<0>();  // in-order: current tile done
    __syncthreads();

    const unsigned short* Ah = &lds[cur * LDS_BUF_US];
    const unsigned short* Al = Ah + A_LO_US;
    const unsigned short* Bh = &lds[cur * LDS_BUF_US + B_HI_US];
    const unsigned short* Bl = &lds[cur * LDS_BUF_US + B_LO_US];

    // A frag (16x32 bf16): lane m=ml, K chunks [hf*8..+7] and [16+hf*8..+7]
    v16bf ah[4], alo[4], bhf[4], blf[4];
#pragma unroll
    for (int tm = 0; tm < 4; ++tm) {
      int r = wm * 64 + tm * 16 + ml;
      ah[tm]  = frag16(*(const U4*)&Ah[r * 40 + hf * 8], *(const U4*)&Ah[r * 40 + 16 + hf * 8]);
      alo[tm] = frag16(*(const U4*)&Al[r * 40 + hf * 8], *(const U4*)&Al[r * 40 + 16 + hf * 8]);
    }
    // B frag (32x16 bf16): lane col=ml, K chunks [hf*8..+7] and [16+hf*8..+7]
#pragma unroll
    for (int tn = 0; tn < 4; ++tn) {
      int c = wn * 64 + tn * 16 + ml;
      bhf[tn] = frag16(*(const U4*)&Bh[c * 40 + hf * 8], *(const U4*)&Bh[c * 40 + 16 + hf * 8]);
      blf[tn] = frag16(*(const U4*)&Bl[c * 40 + hf * 8], *(const U4*)&Bl[c * 40 + 16 + hf * 8]);
    }
#pragma unroll
    for (int tm = 0; tm < 4; ++tm)
#pragma unroll
      for (int tn = 0; tn < 4; ++tn) {
        acc[tm][tn] = __builtin_amdgcn_wmma_f32_16x16x32_bf16(false, ah[tm],  false, bhf[tn], (short)0, acc[tm][tn], false, false);
        acc[tm][tn] = __builtin_amdgcn_wmma_f32_16x16x32_bf16(false, ah[tm],  false, blf[tn], (short)0, acc[tm][tn], false, false);
        acc[tm][tn] = __builtin_amdgcn_wmma_f32_16x16x32_bf16(false, alo[tm], false, bhf[tn], (short)0, acc[tm][tn], false, false);
      }
    __syncthreads();   // all waves done reading `cur` before it is re-staged
    cur ^= 1;
  }

  // epilogue: dist = -2*dot + ||e||^2 (||x||^2 row-constant, irrelevant to argmin)
  float en[4];
#pragma unroll
  for (int tn = 0; tn < 4; ++tn) en[tn] = e_norm[col0 + wn * 64 + tn * 16 + ml];

#pragma unroll
  for (int tm = 0; tm < 4; ++tm) {
#pragma unroll
    for (int r = 0; r < 8; ++r) {
      unsigned long long best = 0xFFFFFFFFFFFFFFFFull;
#pragma unroll
      for (int tn = 0; tn < 4; ++tn) {
        float d = -2.0f * acc[tm][tn][r] + en[tn];
        unsigned cg = (unsigned)(col0 + wn * 64 + tn * 16 + ml);
        unsigned long long key = ((unsigned long long)f2ord(d) << 32) | cg;
        best = best < key ? best : key;
      }
#pragma unroll
      for (int s = 1; s < 16; s <<= 1) {
        unsigned long long o = shfl_xor_u64(best, s);
        best = best < o ? best : o;
      }
      if (ml == 0) {
        int row = row0 + wm * 64 + tm * 16 + hf * 8 + r;
        atomicMin(&minpack[row], best);
      }
    }
  }
}

// ---------------- gather quantize, counts, embed_sum, commit loss ----------------
__global__ void __launch_bounds__(256) vq_gather(
    const float* __restrict__ input, const float* __restrict__ embed,
    const unsigned long long* __restrict__ minpack,
    float* __restrict__ out_q, float* __restrict__ out_ind,
    float* __restrict__ counts, float* __restrict__ embed_sum,
    float* __restrict__ loss_acc) {
  __shared__ float red[256];
  int n = blockIdx.x, t = threadIdx.x;
  unsigned idx = (unsigned)(minpack[n] & 0xFFFFFFFFull);
  float x = input[(size_t)n * DIMS + t];
  float q = embed[(size_t)t * NCODE + idx];
  out_q[(size_t)n * DIMS + t] = q;
  float dd = q - x;
  red[t] = dd * dd;
  atomicAdd(&embed_sum[(size_t)t * NCODE + idx], x);
  __syncthreads();
  for (int s = 128; s > 0; s >>= 1) {
    if (t < s) red[t] += red[t + s];
    __syncthreads();
  }
  if (t == 0) {
    atomicAdd(loss_acc, red[0]);
    out_ind[n] = (float)idx;
    atomicAdd(&counts[idx], 1.0f);
  }
}

// ---------------- EMA cluster size + global sum ----------------
__global__ void __launch_bounds__(256) vq_ema_cs(
    const float* __restrict__ cluster_size, const float* __restrict__ counts,
    float* __restrict__ out_ncs, float* __restrict__ cs_sum) {
  __shared__ float red[256];
  int c = blockIdx.x * 256 + threadIdx.x;
  float ncs = cluster_size[c] * DECAY + counts[c] * OMD;
  out_ncs[c] = ncs;
  red[threadIdx.x] = ncs;
  __syncthreads();
  for (int s = 128; s > 0; s >>= 1) {
    if (threadIdx.x < s) red[threadIdx.x] += red[threadIdx.x + s];
    __syncthreads();
  }
  if (threadIdx.x == 0) atomicAdd(cs_sum, red[0]);
}

// ---------------- finalize: new_embed_avg, new_embed, loss ----------------
__global__ void __launch_bounds__(256) vq_finalize(
    const float* __restrict__ embed_avg, const float* __restrict__ embed_sum,
    const float* __restrict__ out_ncs, const float* __restrict__ cs_sum,
    const float* __restrict__ loss_acc,
    float* __restrict__ out_new_embed, float* __restrict__ out_new_avg,
    float* __restrict__ out_loss) {
  size_t i = (size_t)blockIdx.x * 256 + threadIdx.x;
  int c = (int)(i % NCODE);
  float S = *cs_sum;
  float ncs = out_ncs[c];
  float sm = (ncs + EPS_) / (S + (float)NCODE * EPS_) * S;
  float nea = embed_avg[i] * DECAY + embed_sum[i] * OMD;
  out_new_avg[i] = nea;
  out_new_embed[i] = nea / sm;
  if (i == 0) *out_loss = *loss_acc / (float)((size_t)N_ROWS * DIMS);
}

extern "C" void kernel_launch(void* const* d_in, const int* in_sizes, int n_in,
                              void* d_out, int out_size, void* d_ws, size_t ws_size,
                              hipStream_t stream) {
  (void)in_sizes; (void)n_in; (void)out_size; (void)ws_size;
  const float* input        = (const float*)d_in[0];   // [8,2048,256]
  const float* embed        = (const float*)d_in[1];   // [256,8192]
  const float* cluster_size = (const float*)d_in[2];   // [8192]
  const float* embed_avg    = (const float*)d_in[3];   // [256,8192]

  float* out = (float*)d_out;
  float* out_q    = out;               // 4194304
  float* out_ind  = out + 4194304;     // 16384 (indices as float)
  float* out_loss = out + 4210688;     // 1
  float* out_ne   = out + 4210689;     // 2097152
  float* out_ncs  = out + 6307841;     // 8192
  float* out_nea  = out + 6316033;     // 2097152

  char* ws = (char*)d_ws;
  unsigned short*     x_hi      = (unsigned short*)(ws);
  unsigned short*     x_lo      = (unsigned short*)(ws + 8388608);
  unsigned short*     e_hi_t    = (unsigned short*)(ws + 16777216);  // [C][D]
  unsigned short*     e_lo_t    = (unsigned short*)(ws + 20971520);  // [C][D]
  float*              e_norm    = (float*)(ws + 25165824);
  unsigned long long* minpack   = (unsigned long long*)(ws + 25198592);
  float*              counts    = (float*)(ws + 25329664);
  float*              embed_sum = (float*)(ws + 25362432);
  float*              loss_acc  = (float*)(ws + 33751040);
  float*              cs_sum    = (float*)(ws + 33751044);

  vq_init<<<8192, 256, 0, stream>>>(embed_sum, minpack, counts, loss_acc, cs_sum);
  vq_split<<<16384, 256, 0, stream>>>(input, x_hi, x_lo, (size_t)N_ROWS * DIMS);
  vq_split_t<<<8192, 256, 0, stream>>>(embed, e_hi_t, e_lo_t);
  vq_enorm<<<NCODE / 256, 256, 0, stream>>>(embed, e_norm);

  dim3 g(NCODE / 256, N_ROWS / 128);
  vq_argmin_gemm<<<g, 256, 0, stream>>>(x_hi, x_lo, e_hi_t, e_lo_t, e_norm, minpack);

  vq_gather<<<N_ROWS, 256, 0, stream>>>(input, embed, minpack, out_q, out_ind,
                                        counts, embed_sum, loss_acc);
  vq_ema_cs<<<NCODE / 256, 256, 0, stream>>>(cluster_size, counts, out_ncs, cs_sum);
  vq_finalize<<<(DIMS * NCODE) / 256, 256, 0, stream>>>(embed_avg, embed_sum, out_ncs,
                                                        cs_sum, loss_acc, out_ne,
                                                        out_nea, out_loss);
}
